// Dense_79671643341680
// MI455X (gfx1250) — 8-bit matrix operand order repaired and checked by reading, compiled, not run on hardware
//
#include <hip/hip_runtime.h>
#include <hip/hip_bf16.h>
#include <math.h>

__device__ inline unsigned f32_to_e4m3(float f) {
    unsigned u = __float_as_uint(f);
    unsigned s = (u >> 24) & 0x80u;
    unsigned a = u & 0x7FFFFFFFu;
    if (a > 0x43E00000u) a = 0x43E00000u;
    if (a < 0x3A800000u) return s;
    int e = (int)(a >> 23) - 127;
    unsigned m = a & 0x7FFFFFu;
    unsigned q;
    if (e < -6) {
        m |= 0x800000u;
        int shift = 20 + (-6 - e);
        unsigned half = 1u << (shift - 1);
        unsigned lsb  = (m >> shift) & 1u;
        q = (m + half - 1u + lsb) >> shift;
    } else {
        unsigned half = 1u << 19;
        unsigned r = m + half - 1u + ((m >> 20) & 1u);
        unsigned mant;
        if (r & 0x800000u) { mant = 0u; e += 1; } else { mant = r >> 20; }
        if (e > 8) { e = 8; mant = 6u; }
        q = ((unsigned)(e + 7) << 3) | mant;
    }
    return s | q;
}

__device__ inline float gelu_tanh(float v) {
    float t = tanhf(0.7978845608028654f * (v + 0.044715f * v * v * v));
    return 0.5f * v * (1.0f + t);
}

__global__ void init_kernel(unsigned* ws) {
    ws[0] = 0u;
    ws[1] = 0u;
}

__global__ __launch_bounds__(256) void amax_kernel(const float* __restrict__ x,
                                                   size_t n, unsigned* out) {
    __shared__ float red[256];
    float m = 0.0f;
    for (size_t i = (size_t)blockIdx.x * blockDim.x + threadIdx.x; i < n;
         i += (size_t)gridDim.x * blockDim.x)
        m = fmaxf(m, fabsf(x[i]));
    red[threadIdx.x] = m;
    __syncthreads();
    for (int s = 128; s > 0; s >>= 1) {
        if ((int)threadIdx.x < s)
            red[threadIdx.x] = fmaxf(red[threadIdx.x], red[threadIdx.x + s]);
        __syncthreads();
    }
    if (threadIdx.x == 0) atomicMax(out, __float_as_uint(red[0]));
}

__global__ void scales_kernel(float* wsf) {
    const unsigned* wsu = (const unsigned*)wsf;
    float ax = __uint_as_float(wsu[0]);
    float aw = __uint_as_float(wsu[1]);
    float sx = 448.0f / fmaxf(ax, 1e-12f);
    float sw = 448.0f / fmaxf(aw, 1e-12f);
    wsf[2] = sx;
    wsf[3] = sw;
    wsf[4] = 1.0f / (sx * sw);
}

__global__ __launch_bounds__(256) void quantx_kernel(const float* __restrict__ x,
                                                     unsigned char* __restrict__ q,
                                                     const float* __restrict__ scales,
                                                     size_t n4) {
    size_t i = (size_t)blockIdx.x * blockDim.x + threadIdx.x;
    if (i >= n4) return;
    const float s = scales[2];
    float4 v = ((const float4*)x)[i];
    unsigned packed = f32_to_e4m3(v.x * s)
                    | (f32_to_e4m3(v.y * s) << 8)
                    | (f32_to_e4m3(v.z * s) << 16)
                    | (f32_to_e4m3(v.w * s) << 24);
    ((unsigned*)q)[i] = packed;
}

__global__ __launch_bounds__(256) void quantw_kernel(const float* __restrict__ w,
                                                     unsigned char* __restrict__ bt,
                                                     const float* __restrict__ scales,
                                                     int Kd, int Nd) {
    __shared__ unsigned char tile[32][33];
    const float s = scales[3];
    const int n0 = blockIdx.x * 32;
    const int k0 = blockIdx.y * 32;
    const int tx = threadIdx.x;
    const int ty = threadIdx.y;
    #pragma unroll
    for (int r = ty; r < 32; r += 8) {
        long k = k0 + r, n = n0 + tx;
        tile[r][tx] = (unsigned char)f32_to_e4m3(w[k * (long)Nd + n] * s);
    }
    __syncthreads();
    #pragma unroll
    for (int r = ty; r < 32; r += 8) {
        long n = n0 + r, k = k0 + tx;
        bt[n * (long)Kd + k] = tile[tx][r];
    }
}

typedef __attribute__((ext_vector_type(16))) int   v16i;
typedef __attribute__((ext_vector_type(8)))  float v8f;
typedef __attribute__((ext_vector_type(2)))  int   v2i;
typedef __attribute__((ext_vector_type(4)))  int   v4i;

union Frag16 { v16i v; v2i d2[8]; v4i d4[4]; };

__global__ __launch_bounds__(256) void gemm_fp8_kernel(
        const unsigned char* __restrict__ Aq,
        const unsigned char* __restrict__ Bt,
        const float* __restrict__ bias,
        const float* __restrict__ scales,
        float* __restrict__ out,
        int M, int N, int K) {
    __shared__ alignas(16) unsigned char As[2][128][128];
    __shared__ alignas(16) unsigned char Bs[2][128][128];

    const int t    = threadIdx.x;
    const int lane = t & 31;
    const int wave = t >> 5;
    const int wm   = wave & 3;
    const int wn   = wave >> 2;
    const int lmod = lane & 15;
    const int lhi  = lane >> 4;
    const long mBlk = (long)blockIdx.y * 128;
    const long nBlk = (long)blockIdx.x * 128;

    const int skoff = (t & 7) * 16;
    const int srow0 = t >> 3;
    const unsigned char* aStage = Aq + (mBlk + srow0) * (long)K + skoff;
    const unsigned char* bStage = Bt + (nBlk + srow0) * (long)K + skoff;

    #pragma unroll
    for (int p = 0; p < 4; ++p) {
        const long roff = (long)p * 32 * K;
        v4i av = *(const v4i*)(aStage + roff);
        v4i bv = *(const v4i*)(bStage + roff);
        *(v4i*)&As[0][srow0 + 32 * p][skoff] = av;
        *(v4i*)&Bs[0][srow0 + 32 * p][skoff] = bv;
    }
    __syncthreads();

    v8f acc[2][4] = {};
    const int NK = K >> 7;

    for (int kt = 0; kt < NK; ++kt) {
        const int cur  = kt & 1;
        const bool more = (kt + 1) < NK;

        if (kt + 2 < NK) {
            const long pb = (long)(kt + 2) << 7;
            __builtin_prefetch((const void*)(aStage + pb), 0, 2);
            __builtin_prefetch((const void*)(bStage + pb), 0, 2);
        }

        v4i aReg[4], bReg[4];
        if (more) {
            const long kb = (long)(kt + 1) << 7;
            #pragma unroll
            for (int p = 0; p < 4; ++p) {
                const long roff = (long)p * 32 * K + kb;
                aReg[p] = *(const v4i*)(aStage + roff);
                bReg[p] = *(const v4i*)(bStage + roff);
            }
        }

        Frag16 a[2], b[4];
        #pragma unroll
        for (int i = 0; i < 2; ++i) {
            const unsigned char* pa = &As[cur][wm * 32 + i * 16 + lmod][lhi * 16];
            #pragma unroll
            for (int c = 0; c < 4; ++c) {
                a[i].d4[c] = *(const v4i*)(pa + c * 32);
            }
        }
        #pragma unroll
        for (int j = 0; j < 4; ++j) {
            const unsigned char* pb = &Bs[cur][wn * 64 + j * 16 + lmod][lhi * 16];
            #pragma unroll
            for (int c = 0; c < 4; ++c)
                b[j].d4[c] = *(const v4i*)(pb + c * 32);
        }
        #pragma unroll
        for (int i = 0; i < 2; ++i)
            #pragma unroll
            for (int j = 0; j < 4; ++j)
                acc[i][j] = __builtin_amdgcn_wmma_f32_16x16x128_fp8_fp8(
                    a[i].v, b[j].v, (short)0, acc[i][j], false, false);

        if (more) {
            const int nxt = cur ^ 1;
            #pragma unroll
            for (int p = 0; p < 4; ++p) {
                *(v4i*)&As[nxt][srow0 + 32 * p][skoff] = aReg[p];
                *(v4i*)&Bs[nxt][srow0 + 32 * p][skoff] = bReg[p];
            }
        }
        __syncthreads();
    }

    const float inv = scales[4];
    #pragma unroll
    for (int i = 0; i < 2; ++i) {
        const long row0 = mBlk + wm * 32 + i * 16 + lhi * 8;
        #pragma unroll
        for (int j = 0; j < 4; ++j) {
            const int col = (int)(nBlk + wn * 64 + j * 16 + lmod);
            const float bv = bias[col];
            #pragma unroll
            for (int r = 0; r < 8; ++r) {
                float v = acc[i][j][r] * inv + bv;
                out[(row0 + r) * (long)N + col] = gelu_tanh(v);
            }
        }
    }
}

extern "C" void kernel_launch(void* const* d_in, const int* in_sizes, int n_in,
                              void* d_out, int out_size, void* d_ws, size_t ws_size,
                              hipStream_t stream) {
    const float* x    = (const float*)d_in[0];
    const float* w    = (const float*)d_in[1];
    const float* bias = (const float*)d_in[2];
    const int Nd = in_sizes[2];
    const int Kd = in_sizes[1] / Nd;
    const int Md = in_sizes[0] / Kd;

    unsigned*      wsu = (unsigned*)d_ws;
    float*         wsf = (float*)d_ws;
    unsigned char* xq  = (unsigned char*)d_ws + 64;
    unsigned char* wt  = xq + (size_t)Md * (size_t)Kd;
    float*         out = (float*)d_out;

    init_kernel<<<1, 1, 0, stream>>>(wsu);
    amax_kernel<<<1024, 256, 0, stream>>>(x, (size_t)Md * Kd, wsu + 0);
    amax_kernel<<<2048, 256, 0, stream>>>(w, (size_t)Kd * Nd, wsu + 1);
    scales_kernel<<<1, 1, 0, stream>>>(wsf);

    size_t n4 = (size_t)Md * Kd / 4;
    quantx_kernel<<<(unsigned)((n4 + 255) / 256), 256, 0, stream>>>(x, xq, wsf, n4);
    quantw_kernel<<<dim3(Nd / 32, Kd / 32), dim3(32, 8), 0, stream>>>(w, wt, wsf, Kd, Nd);

    gemm_fp8_kernel<<<dim3(Nd / 128, Md / 128), 256, 0, stream>>>(
        xq, wt, bias, wsf, out, Md, Nd, Kd);
}
